// BridgeDPI_86844238725685
// MI455X (gfx1250) — matrix operand order repaired and checked by reading, not compiled, not run on hardware
//
#include <hip/hip_runtime.h>
#include <hip/hip_bf16.h>
#include <cstdint>
#include <cstddef>

// ---------------------------------------------------------------------------
// BridgeDPI forward for MI455X (gfx1250, wave32, WMMA).
// All dense layers -> one generic bf16 WMMA GEMM (f32 accumulate) with fused
// bias + eval-BatchNorm + ReLU + residual epilogue. Convs / LN / maxpool /
// cosine adjacency are small LDS-cooperative VALU kernels (bandwidth-trivial).
// ---------------------------------------------------------------------------

typedef __bf16 bf16_t;
typedef __attribute__((ext_vector_type(16))) __bf16 v16bf;
typedef __attribute__((ext_vector_type(8)))  __bf16 v8bf;
typedef __attribute__((ext_vector_type(8)))  float  v8f;

#define BN_RS 0.99999500003749969f /* 1/sqrt(1 + 1e-5): eval BN scale factor */

// epilogue flags
enum { EP_BIAS = 1, EP_BN = 2, EP_RELU = 4, EP_RES = 8, EP_OUT_F32 = 16, EP_OUT_BF16 = 32 };

// --------------------------- WMMA fragment loads ---------------------------
// bf16 A (16x32 MxK) layout, ISA 7.12.2: lanes 0-15 hold K[0..8)+K[16..24),
// lanes 16-31 hold K[8..16)+K[24..32) for row M = lane%16.
__device__ __forceinline__ v16bf join8(v8bf lo, v8bf hi) {
    union { v16bf v; v8bf h[2]; } u;
    u.h[0] = lo; u.h[1] = hi;
    return u.v;
}

__device__ __forceinline__ v16bf load_a_frag(const bf16_t* __restrict__ A, int lda,
                                             int row0, int k0, int r, int half) {
    const bf16_t* p = A + (size_t)(row0 + r) * lda + k0 + half * 8;
    return join8(*(const v8bf*)p, *(const v8bf*)(p + 16));
}

// bf16 B (32x16 KxN): lanes 0-15 (col N=lane) hold K[0..8)+K[16..24), lanes 16-31 hold
// K[8..16)+K[24..32) for col N=lane-16 (same K order as A).
// B memory here is W[N,K] row-major (i.e. math B^T), so each lane reads a
// 16B run at k0+8*half and another at k0+16+8*half of its column's K slice.
__device__ __forceinline__ v16bf load_b_frag(const bf16_t* __restrict__ B, int ldb,
                                             int col0, int k0, int r, int half) {
    const bf16_t* p = B + (size_t)(col0 + r) * ldb + k0 + half * 8;
    return join8(*(const v8bf*)p, *(const v8bf*)(p + 16));
}

// --------------------------------- GEMM ------------------------------------
// C[M,N] = act(BN(A[M,K] * W[N,K]^T + bias)) (+ residual), A/W bf16, acc f32.
// Block tile 128(M) x 64(N), 8 waves (4x2), wave tile 32x32 (2x2 WMMA frags).
// Requires: M%128==0, N%64==0, K%32==0 (all call sites satisfy this).
__global__ __launch_bounds__(256) void gemm_bf16_wmma(
    const bf16_t* __restrict__ A, const bf16_t* __restrict__ B,
    const float* __restrict__ bias, const float* __restrict__ bng,
    const float* __restrict__ bnb, const float* __restrict__ res, int ldres,
    float* __restrict__ outF, int ldoF, bf16_t* __restrict__ outH, int ldoH,
    int M, int N, int K, int lda, int ldb, int flags)
{
    const int lane = threadIdx.x & 31;
    const int wave = threadIdx.x >> 5;
    const int wm = wave & 3, wn = wave >> 2;
    const int m_w = blockIdx.y * 128 + wm * 32;
    const int n_w = blockIdx.x * 64 + wn * 32;
    const int r = lane & 15, half = lane >> 4;

    v8f acc[2][2] = {};
    for (int k0 = 0; k0 < K; k0 += 32) {
        v16bf a0 = load_a_frag(A, lda, m_w,      k0, r, half);
        v16bf a1 = load_a_frag(A, lda, m_w + 16, k0, r, half);
        v16bf b0 = load_b_frag(B, ldb, n_w,      k0, r, half);
        v16bf b1 = load_b_frag(B, ldb, n_w + 16, k0, r, half);
        acc[0][0] = __builtin_amdgcn_wmma_f32_16x16x32_bf16(false, a0, false, b0, (short)0, acc[0][0], false, false);
        acc[0][1] = __builtin_amdgcn_wmma_f32_16x16x32_bf16(false, a0, false, b1, (short)0, acc[0][1], false, false);
        acc[1][0] = __builtin_amdgcn_wmma_f32_16x16x32_bf16(false, a1, false, b0, (short)0, acc[1][0], false, false);
        acc[1][1] = __builtin_amdgcn_wmma_f32_16x16x32_bf16(false, a1, false, b1, (short)0, acc[1][1], false, false);
    }

    // C/D layout: VGPR v, lane<16 -> (M = v,   N = lane); lane>=16 -> (M = v+8, N = lane-16)
#pragma unroll
    for (int i = 0; i < 2; ++i) {
#pragma unroll
        for (int j = 0; j < 2; ++j) {
            const int col = n_w + j * 16 + r;
            float bs = (flags & EP_BIAS) ? bias[col] : 0.0f;
            float sc = 1.0f, sh = 0.0f;
            if (flags & EP_BN) { sc = bng[col] * BN_RS; sh = bnb[col]; }
#pragma unroll
            for (int v = 0; v < 8; ++v) {
                const int row = m_w + i * 16 + half * 8 + v;
                float x = acc[i][j][v] + bs;
                if (flags & EP_BN)   x = x * sc + sh;
                if (flags & EP_RELU) x = fmaxf(x, 0.0f);
                if (flags & EP_RES)  x += res[(size_t)row * ldres + col];
                if (flags & EP_OUT_F32)  outF[(size_t)row * ldoF + col] = x;
                if (flags & EP_OUT_BF16) outH[(size_t)row * ldoH + col] = (bf16_t)x;
            }
        }
    }
}

// ---------------------- f32 -> bf16 converters (K-padded) ------------------
__global__ void bn_convert_kernel(const float* __restrict__ x, const float* __restrict__ g,
                                  const float* __restrict__ b, bf16_t* __restrict__ y,
                                  int rows, int Kin, int Kpad) {
    int idx = blockIdx.x * blockDim.x + threadIdx.x;
    if (idx >= rows * Kpad) return;
    int rr = idx / Kpad, k = idx - rr * Kpad;
    float v = 0.0f;
    if (k < Kin) v = x[(size_t)rr * Kin + k] * (g[k] * BN_RS) + b[k];
    y[idx] = (bf16_t)v;
}

__global__ void convert_w_kernel(const float* __restrict__ x, bf16_t* __restrict__ y,
                                 int rows, int Kin, int Kpad) {
    int idx = blockIdx.x * blockDim.x + threadIdx.x;
    if (idx >= rows * Kpad) return;
    int rr = idx / Kpad, k = idx - rr * Kpad;
    y[idx] = (bf16_t)((k < Kin) ? x[(size_t)rr * Kin + k] : 0.0f);
}

// ------------------------ pCNN: one-hot conv == gather ---------------------
// y[b,t,c] = pconv_b[c] + sum_k w[c, ids[b,t+k-12], k]; LN(64)+ReLU+max_t.
// block = one batch row; 4 position-groups x 64 channels.
__global__ __launch_bounds__(256) void pcnn_kernel(
    const int* __restrict__ ids, const float* __restrict__ w,
    const float* __restrict__ bias, const float* __restrict__ lng,
    const float* __restrict__ lnb, bf16_t* __restrict__ pmax)
{
    const int b = blockIdx.x;
    const int c = threadIdx.x & 63;
    const int tg = threadIdx.x >> 6;
    __shared__ int s_ids[512];
    __shared__ float s_acc[4][64];
    for (int i = threadIdx.x; i < 512; i += 256) s_ids[i] = ids[b * 512 + i];
    __syncthreads();
    const float bc = bias[c], gc = lng[c], bt = lnb[c];
    const float* wc = w + c * 1000; // [40,25] slice
    float mx = -1e30f;
    for (int t = tg; t < 512; t += 4) {
        float acc = bc;
#pragma unroll
        for (int k = 0; k < 25; ++k) {
            int pos = t + k - 12;
            if (pos >= 0 && pos < 512) acc += wc[s_ids[pos] * 25 + k];
        }
        s_acc[tg][c] = acc;
        __syncthreads();
        float sum = 0.f, sq = 0.f;
        for (int u = 0; u < 64; ++u) { float v = s_acc[tg][u]; sum += v; sq += v * v; }
        __syncthreads();
        float mean = sum * (1.0f / 64.0f);
        float var  = sq * (1.0f / 64.0f) - mean * mean;
        float val  = (acc - mean) * rsqrtf(var + 1e-5f) * gc + bt;
        mx = fmaxf(mx, fmaxf(val, 0.0f));
    }
    s_acc[tg][c] = mx;
    __syncthreads();
    if (tg == 0) {
        float m = fmaxf(fmaxf(s_acc[0][c], s_acc[1][c]), fmaxf(s_acc[2][c], s_acc[3][c]));
        pmax[b * 64 + c] = (bf16_t)m;
    }
}

// ------------------------------- dCNN --------------------------------------
// y[b,t,c] = bias[c] + sum_{k,i} x[b,t+k-3,i]*w[c,i,k]; LN(64)+ReLU+max_t.
__global__ __launch_bounds__(256) void dcnn_kernel(
    const float* __restrict__ x, const float* __restrict__ w,
    const float* __restrict__ bias, const float* __restrict__ lng,
    const float* __restrict__ lnb, bf16_t* __restrict__ dmax)
{
    const int b = blockIdx.x;
    const int c = threadIdx.x & 63;
    const int tg = threadIdx.x >> 6;
    __shared__ float s_x[128 * 75];
    __shared__ float s_acc[4][64];
    for (int i = threadIdx.x; i < 128 * 75; i += 256) s_x[i] = x[(size_t)b * (128 * 75) + i];
    __syncthreads();
    const float bc = bias[c], gc = lng[c], bt = lnb[c];
    const float* wc = w + c * 525; // [75,7] slice
    float mx = -1e30f;
    for (int t = tg; t < 128; t += 4) {
        float acc = bc;
#pragma unroll
        for (int k = 0; k < 7; ++k) {
            int pos = t + k - 3;
            if (pos >= 0 && pos < 128) {
                const float* xr = &s_x[pos * 75];
                for (int i = 0; i < 75; ++i) acc += xr[i] * wc[i * 7 + k];
            }
        }
        s_acc[tg][c] = acc;
        __syncthreads();
        float sum = 0.f, sq = 0.f;
        for (int u = 0; u < 64; ++u) { float v = s_acc[tg][u]; sum += v; sq += v * v; }
        __syncthreads();
        float mean = sum * (1.0f / 64.0f);
        float var  = sq * (1.0f / 64.0f) - mean * mean;
        float val  = (acc - mean) * rsqrtf(var + 1e-5f) * gc + bt;
        mx = fmaxf(mx, fmaxf(val, 0.0f));
    }
    s_acc[tg][c] = mx;
    __syncthreads();
    if (tg == 0) {
        float m = fmaxf(fmaxf(s_acc[0][c], s_acc[1][c]), fmaxf(s_acc[2][c], s_acc[3][c]));
        dmax[b * 64 + c] = (bf16_t)m;
    }
}

// node[b, 2+i, :] = node_emb[i, :]
__global__ void node_emb_kernel(const float* __restrict__ emb, float* __restrict__ node) {
    int idx = blockIdx.x * blockDim.x + threadIdx.x;
    if (idx >= 512 * 64 * 128) return;
    int k = idx & 127, i = (idx >> 7) & 63, b = idx >> 13;
    node[(size_t)b * (66 * 128) + (2 + i) * 128 + k] = emb[i * 128 + k];
}

// ------------------- cosine adjacency + symmetric normalize ----------------
__global__ __launch_bounds__(256) void adj_kernel(const float* __restrict__ node,
                                                  float* __restrict__ pL) {
    const int b = blockIdx.x;
    __shared__ float s_n[66 * 128];
    __shared__ float s_cos[66 * 66];
    __shared__ float s_d[66];
    const float* nb = node + (size_t)b * 66 * 128;
    for (int i = threadIdx.x; i < 66 * 128; i += 256) s_n[i] = nb[i];
    __syncthreads();
    for (int p = threadIdx.x; p < 66 * 66; p += 256) {
        int i = p / 66, j = p - (p / 66) * 66;
        float dot = 0.f;
        for (int k = 0; k < 128; ++k) dot += s_n[i * 128 + k] * s_n[j * 128 + k];
        s_cos[p] = dot;
    }
    __syncthreads();
    for (int i = threadIdx.x; i < 66; i += 256) s_d[i] = sqrtf(s_cos[i * 66 + i] + 1e-8f);
    __syncthreads();
    for (int p = threadIdx.x; p < 66 * 66; p += 256) {
        int i = p / 66, j = p - i * 66;
        float c = s_cos[p] / (s_d[i] * s_d[j] + 1e-8f);
        c = fmaxf(c, 0.0f);
        if (i == j) c = 1.0f;
        s_cos[p] = c;
    }
    __syncthreads();
    for (int i = threadIdx.x; i < 66; i += 256) {
        float s = 0.f;
        for (int j = 0; j < 66; ++j) s += s_cos[i * 66 + j];
        s_d[i] = rsqrtf(s);
    }
    __syncthreads();
    float* pb = pL + (size_t)b * 66 * 66;
    for (int p = threadIdx.x; p < 66 * 66; p += 256) {
        int i = p / 66, j = p - i * 66;
        pb[p] = s_cos[p] * s_d[i] * s_d[j];
    }
}

// t[b] = pL[b] @ x[b]  ([66,66]@[66,128]); emit bf16 for the WMMA linear.
__global__ __launch_bounds__(256) void gcn_agg_kernel(const float* __restrict__ pL,
                                                      const float* __restrict__ x,
                                                      bf16_t* __restrict__ t) {
    const int b = blockIdx.x;
    __shared__ float s_p[66 * 66];
    __shared__ float s_x[66 * 128];
    const float* pb = pL + (size_t)b * 66 * 66;
    const float* xb = x + (size_t)b * 66 * 128;
    for (int i = threadIdx.x; i < 66 * 66; i += 256) s_p[i] = pb[i];
    for (int i = threadIdx.x; i < 66 * 128; i += 256) s_x[i] = xb[i];
    __syncthreads();
    bf16_t* tb = t + (size_t)b * 66 * 128;
    for (int p = threadIdx.x; p < 66 * 128; p += 256) {
        int i = p >> 7, k = p & 127;
        float acc = 0.f;
        for (int j = 0; j < 66; ++j) acc += s_p[i * 66 + j] * s_x[j * 128 + k];
        tb[p] = (bf16_t)acc;
    }
}

// out[b] = sum_k o[b,0,k]*o[b,1,k]*fc_w[k] + fc_b
__global__ void final_kernel(const float* __restrict__ o, const float* __restrict__ fcw,
                             const float* __restrict__ fcb, float* __restrict__ out) {
    int b = blockIdx.x * blockDim.x + threadIdx.x;
    if (b >= 512) return;
    const float* r0 = o + (size_t)b * 66 * 128;
    const float* r1 = r0 + 128;
    float acc = 0.f;
    for (int k = 0; k < 128; ++k) acc += r0[k] * r1[k] * fcw[k];
    out[b] = acc + fcb[0];
}

// ---------------------------------------------------------------------------
enum {
    IN_AMINO = 0, IN_IDS, IN_FIN, IN_FEA,
    P_NODE_EMB, P_PCONV_W, P_PCONV_B, P_PLN_G, P_PLN_B,
    P_PFC_W, P_PFC_B, P_PFC_BN_G, P_PFC_BN_B,
    P_DCONV_W, P_DCONV_B, P_DLN_G, P_DLN_B,
    P_DFC_W, P_DFC_B, P_DFC_BN_G, P_DFC_BN_B,
    P_CFC_INBN_G, P_CFC_INBN_B, P_CFC_W1, P_CFC_B1, P_CFC_BN1_G, P_CFC_BN1_B,
    P_CFC_W2, P_CFC_B2, P_CFC_BN2_G, P_CFC_BN2_B,
    P_FFC_INBN_G, P_FFC_INBN_B, P_FFC_W1, P_FFC_B1, P_FFC_BN1_G, P_FFC_BN1_B,
    P_FFC_W2, P_FFC_B2, P_FFC_BN2_G, P_FFC_BN2_B,
    P_G_W1, P_G_B1, P_G_BN1_G, P_G_BN1_B,
    P_G_W2, P_G_B2, P_G_BN2_G, P_G_BN2_B,
    P_FC_W, P_FC_B
};

extern "C" void kernel_launch(void* const* d_in, const int* in_sizes, int n_in,
                              void* d_out, int out_size, void* d_ws, size_t ws_size,
                              hipStream_t stream) {
    (void)in_sizes; (void)n_in; (void)out_size; (void)ws_size;
    auto F = [&](int i) { return (const float*)d_in[i]; };
    const int* ids = (const int*)d_in[IN_IDS];

    // ---- workspace carve (256B aligned) ----
    char* ws = (char*)d_ws;
    size_t off = 0;
    auto take = [&](size_t bytes) -> char* {
        char* r = ws + off;
        off = (off + bytes + 255) & ~(size_t)255;
        return r;
    };
    bf16_t* Aam = (bf16_t*)take((size_t)512 * 8448 * 2);   // BN(amino), K-padded
    bf16_t* Wc1 = (bf16_t*)take((size_t)1024 * 8448 * 2);
    bf16_t* Hc1 = (bf16_t*)take((size_t)512 * 1024 * 2);
    bf16_t* Wc2 = (bf16_t*)take((size_t)128 * 1024 * 2);
    bf16_t* Afn = (bf16_t*)take((size_t)512 * 1024 * 2);   // BN(atomFin)
    bf16_t* Wf1 = (bf16_t*)take((size_t)256 * 1024 * 2);
    bf16_t* Hf1 = (bf16_t*)take((size_t)512 * 256 * 2);
    bf16_t* Wf2 = (bf16_t*)take((size_t)128 * 256 * 2);
    bf16_t* Pmx = (bf16_t*)take((size_t)512 * 64 * 2);
    bf16_t* Wp  = (bf16_t*)take((size_t)128 * 64 * 2);
    float*  Xp  = (float*) take((size_t)512 * 128 * 4);
    bf16_t* Dmx = (bf16_t*)take((size_t)512 * 64 * 2);
    bf16_t* Wd  = (bf16_t*)take((size_t)128 * 64 * 2);
    float*  Xd  = (float*) take((size_t)512 * 128 * 4);
    bf16_t* Wg1 = (bf16_t*)take((size_t)128 * 128 * 2);
    bf16_t* Wg2 = (bf16_t*)take((size_t)128 * 128 * 2);
    float*  node = (float*)take((size_t)512 * 66 * 128 * 4);
    float*  h    = (float*)take((size_t)512 * 66 * 128 * 4);
    // safe aliases (regions dead by the time these are written):
    float*  pL = (float*) Wc1;   // 8.9MB <= 17.3MB, Wc1 dead after cfc GEMM1
    bf16_t* t  = (bf16_t*)Aam;   // 8.65MB == 8.65MB, Aam dead after cfc GEMM1
    float*  o  = node;           // node dead after GCN layer-1 residual read

    auto gemm = [&](const bf16_t* A, const bf16_t* B, int M, int N, int K, int lda, int ldb,
                    const float* bias, const float* bng, const float* bnb,
                    const float* res, int ldres, float* oF, int ldoF,
                    bf16_t* oH, int ldoH, int flags) {
        dim3 grid((unsigned)(N / 64), (unsigned)(M / 128));
        gemm_bf16_wmma<<<grid, 256, 0, stream>>>(A, B, bias, bng, bnb, res, ldres,
                                                 oF, ldoF, oH, ldoH, M, N, K, lda, ldb, flags);
    };
    auto cvt = [&](const float* x, bf16_t* y, int rows, int Kin, int Kpad) {
        int tot = rows * Kpad;
        convert_w_kernel<<<(tot + 255) / 256, 256, 0, stream>>>(x, y, rows, Kin, Kpad);
    };

    const int EPF = EP_BIAS | EP_BN | EP_RELU;

    // ---- precision conversion / BN folding ----
    bn_convert_kernel<<<(512 * 8448 + 255) / 256, 256, 0, stream>>>(
        F(IN_AMINO), F(P_CFC_INBN_G), F(P_CFC_INBN_B), Aam, 512, 8420, 8448);
    cvt(F(P_CFC_W1), Wc1, 1024, 8420, 8448);
    bn_convert_kernel<<<(512 * 1024 + 255) / 256, 256, 0, stream>>>(
        F(IN_FIN), F(P_FFC_INBN_G), F(P_FFC_INBN_B), Afn, 512, 1024, 1024);
    cvt(F(P_FFC_W1), Wf1, 256, 1024, 1024);
    cvt(F(P_CFC_W2), Wc2, 128, 1024, 1024);
    cvt(F(P_FFC_W2), Wf2, 128, 256, 256);
    cvt(F(P_PFC_W),  Wp,  128, 64, 64);
    cvt(F(P_DFC_W),  Wd,  128, 64, 64);
    cvt(F(P_G_W1),   Wg1, 128, 128, 128);
    cvt(F(P_G_W2),   Wg2, 128, 128, 128);

    // ---- text CNN branches (conv + LN + ReLU + max over L) ----
    pcnn_kernel<<<512, 256, 0, stream>>>(ids, F(P_PCONV_W), F(P_PCONV_B),
                                         F(P_PLN_G), F(P_PLN_B), Pmx);
    dcnn_kernel<<<512, 256, 0, stream>>>(F(IN_FEA), F(P_DCONV_W), F(P_DCONV_B),
                                         F(P_DLN_G), F(P_DLN_B), Dmx);

    // ---- pFc / dFc: [512,64] x [128,64]^T -> f32 ----
    gemm(Pmx, Wp, 512, 128, 64, 64, 64, F(P_PFC_B), F(P_PFC_BN_G), F(P_PFC_BN_B),
         nullptr, 0, Xp, 128, nullptr, 0, EPF | EP_OUT_F32);
    gemm(Dmx, Wd, 512, 128, 64, 64, 64, F(P_DFC_B), F(P_DFC_BN_G), F(P_DFC_BN_B),
         nullptr, 0, Xd, 128, nullptr, 0, EPF | EP_OUT_F32);

    // ---- cFc MLP: 8448 -> 1024 -> 128, residual Xp, write node[:,0,:] ----
    gemm(Aam, Wc1, 512, 1024, 8448, 8448, 8448, F(P_CFC_B1), F(P_CFC_BN1_G), F(P_CFC_BN1_B),
         nullptr, 0, nullptr, 0, Hc1, 1024, EPF | EP_OUT_BF16);
    gemm(Hc1, Wc2, 512, 128, 1024, 1024, 1024, F(P_CFC_B2), F(P_CFC_BN2_G), F(P_CFC_BN2_B),
         Xp, 128, node + 0, 66 * 128, nullptr, 0, EPF | EP_RES | EP_OUT_F32);

    // ---- fFc MLP: 1024 -> 256 -> 128, residual Xd, write node[:,1,:] ----
    gemm(Afn, Wf1, 512, 256, 1024, 1024, 1024, F(P_FFC_B1), F(P_FFC_BN1_G), F(P_FFC_BN1_B),
         nullptr, 0, nullptr, 0, Hf1, 256, EPF | EP_OUT_BF16);
    gemm(Hf1, Wf2, 512, 128, 256, 256, 256, F(P_FFC_B2), F(P_FFC_BN2_G), F(P_FFC_BN2_B),
         Xd, 128, node + 128, 66 * 128, nullptr, 0, EPF | EP_RES | EP_OUT_F32);

    // ---- bridge nodes broadcast: node[:,2:66,:] ----
    node_emb_kernel<<<(512 * 64 * 128 + 255) / 256, 256, 0, stream>>>(F(P_NODE_EMB), node);

    // ---- normalized cosine adjacency ----
    adj_kernel<<<512, 256, 0, stream>>>(node, pL);

    // ---- GCN layer 1: h = relu(bn(pL@node @ W1^T + b1)) + node ----
    gcn_agg_kernel<<<512, 256, 0, stream>>>(pL, node, t);
    gemm(t, Wg1, 512 * 66, 128, 128, 128, 128, F(P_G_B1), F(P_G_BN1_G), F(P_G_BN1_B),
         node, 128, h, 128, nullptr, 0, EPF | EP_RES | EP_OUT_F32);

    // ---- GCN layer 2: o = relu(bn(pL@h @ W2^T + b2)) + h ----
    gcn_agg_kernel<<<512, 256, 0, stream>>>(pL, h, t);
    gemm(t, Wg2, 512 * 66, 128, 128, 128, 128, F(P_G_B2), F(P_G_BN2_G), F(P_G_BN2_B),
         h, 128, o, 128, nullptr, 0, EPF | EP_RES | EP_OUT_F32);

    // ---- y = (o[:,0,:] * o[:,1,:]) @ fc_w^T + fc_b ----
    final_kernel<<<4, 128, 0, stream>>>(o, F(P_FC_W), F(P_FC_B), (float*)d_out);
}
